// NeuralNetwork_48284022342340
// MI455X (gfx1250) — matrix operand order repaired and checked by reading, compiled, not run on hardware
//
#include <hip/hip_runtime.h>
#include <hip/hip_bf16.h>

// ---------------------------------------------------------------------------
// MoE per-atom MLP energy, routed by symbol_id (E=4 experts).
//   1) bucket atoms by expert (index lists)
//   2) pre-convert features -> f16, pre-transpose+convert W1/W2 -> f16 [E][N][K]
//   3) per-expert f16 WMMA GEMMs (fp32 accum), double-buffered LDS,
//      bias+ReLU fused, f16 activations
//   4) per-atom head dot product, deterministic tree reduction to one float
// Compute-bound: ~137 GFLOP vs ~300 MB traffic (everything L2-resident, 192MB).
// ---------------------------------------------------------------------------

typedef __attribute__((ext_vector_type(16))) _Float16 v16h;
typedef __attribute__((ext_vector_type(8)))  float    v8f;

static constexpr int N_ATOMS = 32768;
static constexpr int DIM     = 1024;   // D == H == 1024
static constexpr int NEXP    = 4;
static constexpr int BM      = 128;    // block tile rows (atoms)
static constexpr int BN      = 128;    // block tile cols
static constexpr int BK      = 32;     // K step (one wmma_f32_16x16x32_f16)
static constexpr int LDA     = BK + 8; // LDS row stride in halves (80B: 16B-aligned, conflict-breaking)
static constexpr int TILES_M = N_ATOMS / BM;  // 256 (worst case per expert)
static constexpr int TILES_N = DIM / BN;      // 8

union FragU { v16h v; uint4 q[2]; _Float16 h[16]; };
union U8    { uint4 q; _Float16 h[8]; };

__device__ inline uint4 cvt8_f32_to_f16(const float4 a, const float4 b) {
  U8 u;
  u.h[0] = (_Float16)a.x; u.h[1] = (_Float16)a.y; u.h[2] = (_Float16)a.z; u.h[3] = (_Float16)a.w;
  u.h[4] = (_Float16)b.x; u.h[5] = (_Float16)b.y; u.h[6] = (_Float16)b.z; u.h[7] = (_Float16)b.w;
  return u.q;
}

// ---------------------------------------------------------------------------
__global__ void zero_counts_kernel(int* __restrict__ counts) {
  if (threadIdx.x < NEXP) counts[threadIdx.x] = 0;
}

__global__ void count_atoms_kernel(const int* __restrict__ sym,
                                   int* __restrict__ counts,
                                   int* __restrict__ idx) {
  int n = blockIdx.x * 256 + threadIdx.x;
  if (n >= N_ATOMS) return;
  int e = sym[n];
  int pos = atomicAdd(&counts[e], 1);
  idx[e * N_ATOMS + pos] = n;
}

// f32 -> f16 elementwise (features). 8 elements / thread, vectorized.
__global__ void cvt_features_kernel(const float* __restrict__ F, _Float16* __restrict__ Fh) {
  size_t i = ((size_t)blockIdx.x * 256 + threadIdx.x) * 8;
  const float4* gp = (const float4*)(F + i);
  *(uint4*)(Fh + i) = cvt8_f32_to_f16(gp[0], gp[1]);
}

// W[e][k][n] f32  ->  Wt[e][n][k] f16 (transpose + convert), 32x32 LDS tiles.
__global__ void transpose_cvt_w_kernel(const float* __restrict__ W, _Float16* __restrict__ Wt) {
  __shared__ _Float16 t[32][33];
  const int e  = blockIdx.z;
  const int k0 = blockIdx.x * 32;
  const int n0 = blockIdx.y * 32;
  const int tx = threadIdx.x & 31;   // along n on read, along k on write (coalesced both ways)
  const int ty = threadIdx.x >> 5;   // 8 rows / pass
  const float* src = W + ((size_t)e * DIM + k0) * DIM + n0;
#pragma unroll
  for (int kk = ty; kk < 32; kk += 8)
    t[kk][tx] = (_Float16)src[(size_t)kk * DIM + tx];
  __syncthreads();
  _Float16* dst = Wt + ((size_t)e * DIM + n0) * DIM + k0;
#pragma unroll
  for (int nn = ty; nn < 32; nn += 8)
    dst[(size_t)nn * DIM + tx] = t[tx][nn];
}

// ---------------------------------------------------------------------------
// One GEMM layer:  Out[g, :] = relu( A[g, :] @ Wt[e]^T + bias[e] ), f16 I/O.
// GATHER=true : A rows indexed through per-expert idx lists (atom ids).
// GATHER=false: A rows are compacted rows (off + slot).
// Grid: NEXP * TILES_M * TILES_N blocks, 256 threads (8 waves, 4x2 wave grid),
// each wave owns 32x64 = 2x4 wmma tiles -> 8 v_wmma per K-step.
// LDS double-buffered: one barrier per K-step, global loads overlap WMMA.
// Tail rows are clamped (not zero-filled): outputs are row-independent and
// invalid rows are masked at the store, so no per-iteration divergence.
// ---------------------------------------------------------------------------
template <bool GATHER>
__global__ __launch_bounds__(256)
void expert_gemm_kernel(const _Float16* __restrict__ A,     // [N_ATOMS][DIM] f16
                        const int*      __restrict__ counts,
                        const int*      __restrict__ idx,
                        const _Float16* __restrict__ Wt,    // [E][N][K] f16 (pre-transposed)
                        const float*    __restrict__ bias,  // [E][DIM]
                        _Float16*       __restrict__ Out)   // [N_ATOMS][DIM] by compact row
{
  const int bx  = blockIdx.x;
  const int e   = bx / (TILES_M * TILES_N);
  const int rem = bx % (TILES_M * TILES_N);
  const int tm  = rem / TILES_N;
  const int tn  = rem % TILES_N;

  const int c0 = counts[0], c1 = counts[1], c2 = counts[2], c3 = counts[3];
  const int cntArr[NEXP] = {c0, c1, c2, c3};
  const int cnt = cntArr[e];
  if (tm * BM >= cnt) return;                          // empty tile -> early out
  int off = 0;                                         // exclusive prefix of counts
  if (e > 0) off += c0;
  if (e > 1) off += c1;
  if (e > 2) off += c2;

  __shared__ alignas(16) _Float16 As[2][BM * LDA];     // ping-pong 128 x 32 (padded)
  __shared__ alignas(16) _Float16 Bs[2][BN * LDA];

  const int tid  = threadIdx.x;
  const int lane = tid & 31;
  const int wid  = tid >> 5;
  const int wm   = wid >> 1;       // 0..3 -> rows  [32*wm, 32*wm+32)
  const int wn   = wid & 1;        // 0..1 -> cols  [64*wn, 64*wn+64)
  const int l16  = lane & 15;
  const int lhi  = lane >> 4;      // lane group 0/1

  // ---- A staging: 2 threads per row, 16 halves (2 x b128) each ------------
  const int arow  = tid >> 1;              // 0..127
  const int ahalf = tid & 1;               // k sub-chunk of 16 halves
  const int aslot = tm * BM + arow;
  const int sslot = aslot < cnt ? aslot : (cnt - 1);   // clamped (see note above)
  size_t asrc;
  if (GATHER) asrc = (size_t)idx[e * N_ATOMS + sslot];
  else        asrc = (size_t)(off + sslot);
  const _Float16* abase = &A[asrc * DIM + ahalf * 16];
  const int aoffs = arow * LDA + ahalf * 16;

  // ---- B staging: 2 threads per n-row of Wt tile, 16 halves each ----------
  const int bn    = tid >> 1;              // 0..127
  const int bhalf = tid & 1;
  const _Float16* bbase = &Wt[((size_t)e * DIM + (size_t)(tn * BN + bn)) * DIM + bhalf * 16];
  const int boffs = bn * LDA + bhalf * 16;

  v8f acc[2][4] = {};

  // prologue: stage tile k0=0 into buffer 0
  {
    const uint4* ap = (const uint4*)abase;
    const uint4* bp = (const uint4*)bbase;
    const uint4 a0 = ap[0], a1 = ap[1], b0 = bp[0], b1 = bp[1];
    *(uint4*)&As[0][aoffs]     = a0;
    *(uint4*)&As[0][aoffs + 8] = a1;
    *(uint4*)&Bs[0][boffs]     = b0;
    *(uint4*)&Bs[0][boffs + 8] = b1;
  }
  __syncthreads();

  int buf = 0;
  for (int k0 = 0; k0 < DIM; k0 += BK, buf ^= 1) {
    const bool more = (k0 + BK) < DIM;

    // issue next tile's global loads first (overlap with WMMA below)
    uint4 na0, na1, nb0, nb1;
    if (more) {
      const uint4* ap = (const uint4*)(abase + k0 + BK);
      const uint4* bp = (const uint4*)(bbase + k0 + BK);
      na0 = ap[0]; na1 = ap[1];
      nb0 = bp[0]; nb1 = bp[1];
    }

    // fragments per ISA VGPR layout, from current buffer
    const _Float16* __restrict__ Ac = As[buf];
    const _Float16* __restrict__ Bc = Bs[buf];
    v16h afrag[2];
#pragma unroll
    for (int ti = 0; ti < 2; ++ti) {
      const int r = wm * 32 + ti * 16 + l16;
      FragU u;
      u.q[0] = *(const uint4*)&Ac[r * LDA + lhi * 8];        // K 0-7  / 8-15
      u.q[1] = *(const uint4*)&Ac[r * LDA + lhi * 8 + 16];   // K 16-23/ 24-31
      afrag[ti] = u.v;
    }
    v16h bfrag[4];
#pragma unroll
    for (int tj = 0; tj < 4; ++tj) {
      const int c = wn * 64 + tj * 16 + l16;
      FragU u;
      u.q[0] = *(const uint4*)&Bc[c * LDA + lhi * 8];        // K 0-7  / 8-15
      u.q[1] = *(const uint4*)&Bc[c * LDA + lhi * 8 + 16];   // K 16-23/ 24-31
      bfrag[tj] = u.v;
    }
#pragma unroll
    for (int ti = 0; ti < 2; ++ti)
#pragma unroll
      for (int tj = 0; tj < 4; ++tj)
        acc[ti][tj] = __builtin_amdgcn_wmma_f32_16x16x32_f16(
            false, afrag[ti], false, bfrag[tj], (short)0, acc[ti][tj], false, false);

    // stage next tile into the other buffer (written this iter, read next iter)
    if (more) {
      _Float16* An = As[buf ^ 1];
      _Float16* Bn = Bs[buf ^ 1];
      *(uint4*)&An[aoffs]     = na0;
      *(uint4*)&An[aoffs + 8] = na1;
      *(uint4*)&Bn[boffs]     = nb0;
      *(uint4*)&Bn[boffs + 8] = nb1;
    }
    __syncthreads();   // single barrier per K-step
  }

  // epilogue: bias + ReLU, store f16 (C/D layout: vgpr i -> M = i + 8*lhi)
#pragma unroll
  for (int tj = 0; tj < 4; ++tj) {
    const int gcol = tn * BN + wn * 64 + tj * 16 + l16;
    const float bv = bias[e * DIM + gcol];
#pragma unroll
    for (int ti = 0; ti < 2; ++ti) {
#pragma unroll
      for (int i = 0; i < 8; ++i) {
        const int rrow = wm * 32 + ti * 16 + lhi * 8 + i;
        const int s = tm * BM + rrow;
        if (s < cnt) {
          float v = acc[ti][tj][i] + bv;
          v = v > 0.f ? v : 0.f;
          Out[(size_t)(off + s) * DIM + gcol] = (_Float16)v;
        }
      }
    }
  }
}

// ---------------------------------------------------------------------------
// Head: y = h2 . W3[e] + b3[e];  energy[atom] = slope[e]*y + intercept[e].
// One wave per compacted row; 8 rows per 256-thread block.
// ---------------------------------------------------------------------------
__global__ __launch_bounds__(256)
void head_kernel(const _Float16* __restrict__ H2,
                 const int*      __restrict__ counts,
                 const int*      __restrict__ idx,
                 const float*    __restrict__ W3,     // [E][1024]
                 const float*    __restrict__ b3,     // [E]
                 const float*    __restrict__ slope,  // [E]
                 const float*    __restrict__ inter,  // [E]
                 float*          __restrict__ energies)
{
  const int lane = threadIdx.x & 31;
  const int wid  = threadIdx.x >> 5;
  const int g = blockIdx.x * 8 + wid;
  if (g >= N_ATOMS) return;

  const int c0 = counts[0], c1 = counts[1], c2 = counts[2];
  int e, off;
  if      (g < c0)           { e = 0; off = 0; }
  else if (g < c0 + c1)      { e = 1; off = c0; }
  else if (g < c0 + c1 + c2) { e = 2; off = c0 + c1; }
  else                       { e = 3; off = c0 + c1 + c2; }
  const int atom = idx[e * N_ATOMS + (g - off)];

  const _Float16* h = &H2[(size_t)g * DIM];
  const float*    w = &W3[(size_t)e * DIM];

  float sum = 0.f;
#pragma unroll
  for (int j = 0; j < 4; ++j) {
    const int kb = (lane + 32 * j) * 8;     // 8 contiguous halves per lane
    U8 u; u.q = *(const uint4*)&h[kb];
    const float4 w0 = *(const float4*)&w[kb];
    const float4 w1 = *(const float4*)&w[kb + 4];
    sum += (float)u.h[0] * w0.x + (float)u.h[1] * w0.y +
           (float)u.h[2] * w0.z + (float)u.h[3] * w0.w +
           (float)u.h[4] * w1.x + (float)u.h[5] * w1.y +
           (float)u.h[6] * w1.z + (float)u.h[7] * w1.w;
  }
#pragma unroll
  for (int m = 16; m >= 1; m >>= 1) sum += __shfl_xor(sum, m, 32);

  if (lane == 0) {
    const float y = sum + b3[e];
    energies[atom] = slope[e] * y + inter[e];
  }
}

// ---------------------------------------------------------------------------
// Deterministic final reduction (fixed strided order + fixed tree).
// ---------------------------------------------------------------------------
__global__ void reduce_kernel(const float* __restrict__ energies, float* __restrict__ out) {
  __shared__ float sh[1024];
  float s = 0.f;
  for (int i = threadIdx.x; i < N_ATOMS; i += 1024) s += energies[i];
  sh[threadIdx.x] = s;
  __syncthreads();
  for (int m = 512; m > 0; m >>= 1) {
    if ((int)threadIdx.x < m) sh[threadIdx.x] += sh[threadIdx.x + m];
    __syncthreads();
  }
  if (threadIdx.x == 0) out[0] = sh[0];
}

// ---------------------------------------------------------------------------
extern "C" void kernel_launch(void* const* d_in, const int* in_sizes, int n_in,
                              void* d_out, int out_size, void* d_ws, size_t ws_size,
                              hipStream_t stream) {
  (void)in_sizes; (void)n_in; (void)out_size; (void)ws_size;

  const float* features = (const float*)d_in[0];
  const int*   sym      = (const int*)  d_in[1];
  const float* W1       = (const float*)d_in[2];
  const float* b1       = (const float*)d_in[3];
  const float* W2       = (const float*)d_in[4];
  const float* b2       = (const float*)d_in[5];
  const float* W3       = (const float*)d_in[6];
  const float* b3       = (const float*)d_in[7];
  const float* slope    = (const float*)d_in[8];
  const float* inter    = (const float*)d_in[9];
  float* out = (float*)d_out;

  // Workspace layout (256B-aligned offsets):
  char* ws = (char*)d_ws;
  size_t o = 0;
  int*      counts   = (int*)(ws + o);            o += 256;
  int*      idx      = (int*)(ws + o);            o += (size_t)NEXP * N_ATOMS * 4;   // 512 KB
  float*    energies = (float*)(ws + o);          o += (size_t)N_ATOMS * 4;          // 128 KB
  _Float16* feat16   = (_Float16*)(ws + o);       o += (size_t)N_ATOMS * DIM * 2;    //  64 MB
  _Float16* W1t      = (_Float16*)(ws + o);       o += (size_t)NEXP * DIM * DIM * 2; //   8 MB
  _Float16* W2t      = (_Float16*)(ws + o);       o += (size_t)NEXP * DIM * DIM * 2; //   8 MB
  _Float16* H1       = (_Float16*)(ws + o);       o += (size_t)N_ATOMS * DIM * 2;    //  64 MB
  _Float16* H2       = feat16;  // alias: features (f16) dead after layer 1

  const int gemm_blocks = NEXP * TILES_M * TILES_N;  // 8192 (early-exit on empty tiles)

  zero_counts_kernel<<<1, 32, 0, stream>>>(counts);
  count_atoms_kernel<<<N_ATOMS / 256, 256, 0, stream>>>(sym, counts, idx);
  cvt_features_kernel<<<(N_ATOMS * (DIM / 8)) / 256, 256, 0, stream>>>(features, feat16);
  transpose_cvt_w_kernel<<<dim3(DIM / 32, DIM / 32, NEXP), 256, 0, stream>>>(W1, W1t);
  transpose_cvt_w_kernel<<<dim3(DIM / 32, DIM / 32, NEXP), 256, 0, stream>>>(W2, W2t);
  expert_gemm_kernel<true ><<<gemm_blocks, 256, 0, stream>>>(feat16, counts, idx, W1t, b1, H1);
  expert_gemm_kernel<false><<<gemm_blocks, 256, 0, stream>>>(H1,     counts, idx, W2t, b2, H2);
  head_kernel<<<N_ATOMS / 8, 256, 0, stream>>>(H2, counts, idx, W3, b3, slope, inter, energies);
  reduce_kernel<<<1, 1024, 0, stream>>>(energies, out);
}
